// HAN_32495722562262
// MI455X (gfx1250) — matrix operand order repaired and checked by reading, not compiled, not run on hardware
//
#include <hip/hip_runtime.h>
#include <hip/hip_bf16.h>
#include <cstddef>

// HAN (2-layer heterogeneous attention network) for MI455X / gfx1250.
// GEMMs: LDS-staged split-f16 WMMA (v_wmma_f32_16x16x32_f16 x3 per tile per K-step,
//        fp32 accumulate -> ~fp32 accuracy), coalesced b128 global loads,
//        fragment reads as contiguous ds_load_b128 matching the ISA VGPR layouts.
// Edge softmax/aggregation: fp32 memory-bound path (HBM roofline dominant).
// Layer 2 computes only the author-destination metapaths (output is x2['author']).

#define NA 20000
#define NP 30000
#define NE 200000
#define NH 4

typedef __attribute__((ext_vector_type(16))) _Float16 v16h;
typedef __attribute__((ext_vector_type(8)))  _Float16 v8h;
typedef __attribute__((ext_vector_type(4)))  _Float16 v4h;
typedef __attribute__((ext_vector_type(8)))  float    v8f;
typedef __attribute__((ext_vector_type(4)))  float    vf4;

// ---- order-preserving float <-> uint map for atomic segment-max ----
__device__ __forceinline__ unsigned ford(float f) {
  unsigned b = __float_as_uint(f);
  return (b & 0x80000000u) ? ~b : (b | 0x80000000u);
}
__device__ __forceinline__ float funord(unsigned u) {
  unsigned b = (u & 0x80000000u) ? (u & 0x7fffffffu) : ~u;
  return __uint_as_float(b);
}

// ================= LDS-staged split-f16 WMMA GEMM =================
// C[MxN] = A[MxK] @ W[KxN] + bias[N].  K % 32 == 0, N % 64 == 0, any M.
// Block: 128 threads = 4 waves; block tile 64(M) x 64(N); wave tile 16(M) x 64(N).
#define LDP 40   // LDS row pitch in halves (80B: 16B-aligned, bank-spreading)

__global__ void __launch_bounds__(128)
k_gemm_wmma(const float* __restrict__ A, const float* __restrict__ W,
            const float* __restrict__ bias, float* __restrict__ C,
            int M, int K, int N) {
  __shared__ _Float16 Ah[64 * LDP];   // A hi halves  [m][k]
  __shared__ _Float16 Al[64 * LDP];   // A lo halves  [m][k]
  __shared__ _Float16 Bh[64 * LDP];   // B hi halves  [n][k] (transposed)
  __shared__ _Float16 Bl[64 * LDP];   // B lo halves  [n][k]

  const int tid  = threadIdx.x;
  const int lane = tid & 31;
  const int wave = tid >> 5;
  const int l16  = lane & 15;
  const int m0   = blockIdx.y * 64;
  const int n0   = blockIdx.x * 64;
  const int kA   = (lane < 16) ? 0 : 8;    // ISA 16-bit A fragment K base
  const int kB   = (lane < 16) ? 0 : 8;    // ISA 16-bit B fragment K base

  v8f acc0 = {}, acc1 = {}, acc2 = {}, acc3 = {};

  for (int k0 = 0; k0 < K; k0 += 32) {
    __syncthreads();
    // ---- stage A: 64 x 32 f32, coalesced b128, split to hi/lo halves ----
#pragma unroll
    for (int it = 0; it < 4; ++it) {
      int r = (tid >> 3) + it * 16;            // 0..63
      int c = (tid & 7) * 4;                   // 0..28
      vf4 x = {0.f, 0.f, 0.f, 0.f};
      if (m0 + r < M) x = *(const vf4*)(A + (size_t)(m0 + r) * K + k0 + c);
      v4h hi, lo;
#pragma unroll
      for (int j = 0; j < 4; ++j) {
        _Float16 h = (_Float16)x[j];
        hi[j] = h;
        lo[j] = (_Float16)(x[j] - (float)h);
      }
      *(v4h*)&Ah[r * LDP + c] = hi;
      *(v4h*)&Al[r * LDP + c] = lo;
    }
    // ---- stage B: 32 x 64 f32, coalesced b128, store transposed [n][k] ----
#pragma unroll
    for (int it = 0; it < 4; ++it) {
      int kk = (tid >> 4) + it * 8;            // 0..31
      int c  = (tid & 15) * 4;                 // 0..60
      vf4 x = *(const vf4*)(W + (size_t)(k0 + kk) * N + n0 + c);
#pragma unroll
      for (int j = 0; j < 4; ++j) {
        _Float16 h = (_Float16)x[j];
        Bh[(c + j) * LDP + kk] = h;
        Bl[(c + j) * LDP + kk] = (_Float16)(x[j] - (float)h);
      }
    }
    __syncthreads();

    // ---- A fragments: lane row m = wave*16+l16; K {kA..kA+7} ++ {16+kA..23+kA} ----
    const _Float16* Arh = &Ah[(wave * 16 + l16) * LDP];
    const _Float16* Arl = &Al[(wave * 16 + l16) * LDP];
    v8h a0 = *(const v8h*)(Arh + kA);
    v8h a1 = *(const v8h*)(Arh + 16 + kA);
    v16h ah = __builtin_shufflevector(a0, a1, 0,1,2,3,4,5,6,7,8,9,10,11,12,13,14,15);
    v8h c0 = *(const v8h*)(Arl + kA);
    v8h c1 = *(const v8h*)(Arl + 16 + kA);
    v16h al = __builtin_shufflevector(c0, c1, 0,1,2,3,4,5,6,7,8,9,10,11,12,13,14,15);

    // ---- 4 N-subtiles: B fragment = K {kB..kB+7} ++ {16+kB..23+kB} of column n ----
#pragma unroll
    for (int j = 0; j < 4; ++j) {
      const _Float16* Brh = &Bh[(j * 16 + l16) * LDP + kB];
      const _Float16* Brl = &Bl[(j * 16 + l16) * LDP + kB];
      v8h b0 = *(const v8h*)(Brh);
      v8h b1 = *(const v8h*)(Brh + 16);
      v16h bh = __builtin_shufflevector(b0, b1, 0,1,2,3,4,5,6,7,8,9,10,11,12,13,14,15);
      v8h d0 = *(const v8h*)(Brl);
      v8h d1 = *(const v8h*)(Brl + 16);
      v16h bl = __builtin_shufflevector(d0, d1, 0,1,2,3,4,5,6,7,8,9,10,11,12,13,14,15);
      v8f acc = (j == 0) ? acc0 : (j == 1) ? acc1 : (j == 2) ? acc2 : acc3;
      // split-f16: A*B ~= Ah*Bh + Ah*Bl + Al*Bh (fp32 accumulate)
      acc = __builtin_amdgcn_wmma_f32_16x16x32_f16(false, ah, false, bh, (short)0, acc, false, false);
      acc = __builtin_amdgcn_wmma_f32_16x16x32_f16(false, ah, false, bl, (short)0, acc, false, false);
      acc = __builtin_amdgcn_wmma_f32_16x16x32_f16(false, al, false, bh, (short)0, acc, false, false);
      if (j == 0) acc0 = acc; else if (j == 1) acc1 = acc; else if (j == 2) acc2 = acc; else acc3 = acc;
    }
  }

  // ---- epilogue: C layout lane = N%16, rows (lane<16?0:8)+r ----
  const int rbase = m0 + wave * 16 + ((lane < 16) ? 0 : 8);
#pragma unroll
  for (int j = 0; j < 4; ++j) {
    const v8f acc = (j == 0) ? acc0 : (j == 1) ? acc1 : (j == 2) ? acc2 : acc3;
    const int ncol = n0 + j * 16 + l16;
    const float bv = bias[ncol];
#pragma unroll
    for (int r = 0; r < 8; ++r) {
      const int row = rbase + r;
      if (row < M) C[(size_t)row * N + ncol] = acc[r] + bv;
    }
  }
}

// ---- per-node attention pre-scores: out[n,h] = sum_d x[n,h,d]*att[h,d] ----
__global__ void k_node_scores(const float* __restrict__ x, const float* __restrict__ att,
                              float* __restrict__ out, int Nn, int Dh) {
  int idx = blockIdx.x * blockDim.x + threadIdx.x;
  if (idx >= Nn * NH) return;
  int n = idx / NH, h = idx % NH;
  const float* xp = x + (size_t)n * NH * Dh + h * Dh;
  const float* ap = att + h * Dh;
  float s = 0.f;
  for (int d = 0; d < Dh; ++d) s += xp[d] * ap[d];
  out[idx] = s;
}

__global__ void k_fill(float* p, float v, size_t n) {
  size_t i = (size_t)blockIdx.x * blockDim.x + threadIdx.x;
  if (i < n) p[i] = v;
}

__global__ void k_relu(float* p, size_t n) {
  size_t i = (size_t)blockIdx.x * blockDim.x + threadIdx.x;
  if (i < n) p[i] = fmaxf(p[i], 0.f);
}

// ---- per-edge leaky-relu logits + atomic segment max ----
__global__ void k_edge_logits(const int* __restrict__ row, const int* __restrict__ col,
                              const float* __restrict__ ssrc, const float* __restrict__ sdst,
                              float* __restrict__ abuf, unsigned* __restrict__ amax) {
  int e = blockIdx.x * blockDim.x + threadIdx.x;
  if (e >= NE) return;
  int r = row[e], c = col[e];
#pragma unroll
  for (int h = 0; h < NH; ++h) {
    float a = ssrc[r * NH + h] + sdst[c * NH + h];
    a = (a < 0.f) ? 0.2f * a : a;            // leaky_relu, slope 0.2
    abuf[(size_t)e * NH + h] = a;
    atomicMax(&amax[c * NH + h], ford(a));
  }
}

// ---- ea = exp(a - segmax); atomic segment sum ----
__global__ void k_edge_exp(const int* __restrict__ col, float* __restrict__ abuf,
                           const unsigned* __restrict__ amax, float* __restrict__ denom) {
  int e = blockIdx.x * blockDim.x + threadIdx.x;
  if (e >= NE) return;
  int c = col[e];
#pragma unroll
  for (int h = 0; h < NH; ++h) {
    float m  = funord(amax[c * NH + h]);
    float ea = __expf(abuf[(size_t)e * NH + h] - m);
    abuf[(size_t)e * NH + h] = ea;
    atomicAdd(&denom[c * NH + h], ea);
  }
}

// ---- weighted scatter-sum: agg[col,f] += xs[row,f] * alpha[e, f/(F/H)] ----
__global__ void k_edge_agg(const int* __restrict__ row, const int* __restrict__ col,
                           const float* __restrict__ xs, const float* __restrict__ ea,
                           const float* __restrict__ denom, float* __restrict__ agg, int F) {
  int idx = blockIdx.x * blockDim.x + threadIdx.x;
  int nq = F >> 2;
  if (idx >= NE * nq) return;
  int e  = idx / nq;
  int f  = (idx - e * nq) * 4;
  int h  = f / (F / NH);
  int c  = col[e];
  float alpha = ea[(size_t)e * NH + h] / (denom[c * NH + h] + 1e-16f);
  const float* xp = xs + (size_t)row[e] * F + f;
  float* op = agg + (size_t)c * F + f;
  atomicAdd(op + 0, xp[0] * alpha);
  atomicAdd(op + 1, xp[1] * alpha);
  atomicAdd(op + 2, xp[2] * alpha);
  atomicAdd(op + 3, xp[3] * alpha);
}

// ---- semantic score partial: score += sum_n q . tanh(t[n,:]) (block-reduced) ----
__global__ void k_sem_score(const float* __restrict__ t, const float* __restrict__ q,
                            float* __restrict__ score, int Nn, int F) {
  __shared__ float sh[256];
  int n = blockIdx.x * blockDim.x + threadIdx.x;
  float s = 0.f;
  if (n < Nn) {
    const float* tp = t + (size_t)n * F;
    for (int f = 0; f < F; ++f) s += q[f] * tanhf(tp[f]);
  }
  sh[threadIdx.x] = s;
  __syncthreads();
  for (int off = 128; off > 0; off >>= 1) {
    if ((int)threadIdx.x < off) sh[threadIdx.x] += sh[threadIdx.x + off];
    __syncthreads();
  }
  if (threadIdx.x == 0) atomicAdd(score, sh[0]);
}

__global__ void k_softmax2(const float* __restrict__ score, float* __restrict__ attn, float invN) {
  float s0 = score[0] * invN, s1 = score[1] * invN;
  float m = fmaxf(s0, s1);
  float e0 = __expf(s0 - m), e1 = __expf(s1 - m);
  float d = e0 + e1;
  attn[0] = e0 / d;
  attn[1] = e1 / d;
}

__global__ void k_combine2(const float* __restrict__ a, const float* __restrict__ b,
                           const float* __restrict__ attn, float* __restrict__ out, size_t n) {
  size_t i = (size_t)blockIdx.x * blockDim.x + threadIdx.x;
  if (i < n) out[i] = attn[0] * a[i] + attn[1] * b[i];
}

extern "C" void kernel_launch(void* const* d_in, const int* in_sizes, int n_in,
                              void* d_out, int out_size, void* d_ws, size_t ws_size,
                              hipStream_t stream) {
  (void)in_sizes; (void)n_in; (void)out_size; (void)ws_size;

  const float* x_a    = (const float*)d_in[0];
  const float* x_p    = (const float*)d_in[1];
  const int*   src_ap = (const int*)d_in[2];
  const int*   dst_ap = (const int*)d_in[3];
  const int*   src_pa = (const int*)d_in[4];
  const int*   dst_pa = (const int*)d_in[5];
  const int*   src_pp = (const int*)d_in[6];
  const int*   dst_pp = (const int*)d_in[7];
  const int*   src_aa = (const int*)d_in[8];
  const int*   dst_aa = (const int*)d_in[9];
  const float* W1_a = (const float*)d_in[10]; const float* b1_a = (const float*)d_in[11];
  const float* W1_p = (const float*)d_in[12]; const float* b1_p = (const float*)d_in[13];
  const float* att1_src = (const float*)d_in[14];
  const float* att1_dst = (const float*)d_in[15];
  const float* q1   = (const float*)d_in[16];
  const float* k1_W = (const float*)d_in[17]; const float* k1_b = (const float*)d_in[18];
  const float* W2_a = (const float*)d_in[19]; const float* b2_a = (const float*)d_in[20];
  const float* W2_p = (const float*)d_in[21]; const float* b2_p = (const float*)d_in[22];
  const float* att2_src = (const float*)d_in[23];
  const float* att2_dst = (const float*)d_in[24];
  const float* q2   = (const float*)d_in[25];
  const float* k2_W = (const float*)d_in[26]; const float* k2_b = (const float*)d_in[27];

  // -------- workspace layout (floats), with explicit reuse --------
  float* ws = (float*)d_ws;
  size_t off = 0;
  auto alloc = [&](size_t nf) { float* p = ws + off; off += nf; return p; };

  float* h1_a = alloc((size_t)NA * 256);      // layer-1 author features; reused as res1_a
  float* h1_p = alloc((size_t)NP * 256);      // layer-1 paper features;  reused as res1_p
  float* aggB = alloc((size_t)(2 * NP + 2 * NA) * 256);   // layer-1 aggregation region
  float* agg_ap = aggB;
  float* agg_pa = agg_ap + (size_t)NP * 256;
  float* agg_pp = agg_pa + (size_t)NA * 256;
  float* agg_aa = agg_pp + (size_t)NP * 256;
  // layer-2 buffers alias the (then dead) layer-1 aggregation region
  float* h2_a    = aggB;
  float* h2_p    = h2_a + (size_t)NA * 128;
  float* agg2_pa = h2_p + (size_t)NP * 128;
  float* agg2_aa = agg2_pa + (size_t)NA * 128;
  float* tbuf  = alloc((size_t)NP * 256);     // semantic GEMM temp
  float* abuf  = alloc((size_t)NE * NH);      // edge logits -> exp weights
  float* ssrc  = alloc((size_t)NP * NH);
  float* sdst  = alloc((size_t)NP * NH);
  float* amax  = alloc((size_t)NP * NH);      // ordered-uint segment max
  float* denom = alloc((size_t)NP * NH);
  float* score = alloc(8);                    // [s0, s1, attn0, attn1, ...]
  float* attn  = score + 2;

  auto cdiv = [](long a, long b) { return (a + b - 1) / b; };
  auto gemm = [&](const float* A, const float* W, const float* b, float* C, int M, int K, int N) {
    dim3 g((unsigned)(N / 64), (unsigned)cdiv(M, 64));
    k_gemm_wmma<<<g, 128, 0, stream>>>(A, W, b, C, M, K, N);
  };
  auto fill = [&](float* p, float v, size_t n) {
    k_fill<<<(unsigned)cdiv((long)n, 256), 256, 0, stream>>>(p, v, n);
  };
  auto run_edge = [&](const int* row, const int* col, const float* hs, const float* hd,
                      const float* attS, const float* attD, int Ns, int Nd, int F, float* agg) {
    int Dh = F / NH;
    k_node_scores<<<(unsigned)cdiv((long)Ns * NH, 256), 256, 0, stream>>>(hs, attS, ssrc, Ns, Dh);
    k_node_scores<<<(unsigned)cdiv((long)Nd * NH, 256), 256, 0, stream>>>(hd, attD, sdst, Nd, Dh);
    fill(amax, 0.0f, (size_t)Nd * NH);        // bits==0 is minimum of ordered encoding
    fill(denom, 0.0f, (size_t)Nd * NH);
    k_edge_logits<<<(unsigned)cdiv(NE, 256), 256, 0, stream>>>(row, col, ssrc, sdst, abuf,
                                                               (unsigned*)amax);
    k_edge_exp<<<(unsigned)cdiv(NE, 256), 256, 0, stream>>>(col, abuf, (const unsigned*)amax, denom);
    fill(agg, 0.0f, (size_t)Nd * F);
    long nthr = (long)NE * (F / 4);
    k_edge_agg<<<(unsigned)cdiv(nthr, 256), 256, 0, stream>>>(row, col, hs, abuf, denom, agg, F);
    k_relu<<<(unsigned)cdiv((long)Nd * F, 256), 256, 0, stream>>>(agg, (size_t)Nd * F);
  };
  auto run_sem = [&](const float* s0b, const float* s1b, const float* kW, const float* kb,
                     const float* q, int Nn, int F, float* res) {
    fill(score, 0.0f, 2);
    gemm(s0b, kW, kb, tbuf, Nn, F, F);
    k_sem_score<<<(unsigned)cdiv(Nn, 256), 256, 0, stream>>>(tbuf, q, score + 0, Nn, F);
    gemm(s1b, kW, kb, tbuf, Nn, F, F);
    k_sem_score<<<(unsigned)cdiv(Nn, 256), 256, 0, stream>>>(tbuf, q, score + 1, Nn, F);
    k_softmax2<<<1, 1, 0, stream>>>(score, attn, 1.0f / (float)Nn);
    k_combine2<<<(unsigned)cdiv((long)Nn * F, 256), 256, 0, stream>>>(s0b, s1b, attn, res,
                                                                      (size_t)Nn * F);
  };

  // ---------------- layer 1 ----------------
  gemm(x_a, W1_a, b1_a, h1_a, NA, 256, 256);
  gemm(x_p, W1_p, b1_p, h1_p, NP, 256, 256);

  run_edge(src_ap, dst_ap, h1_a, h1_p, att1_src + 0 * 256, att1_dst + 0 * 256, NA, NP, 256, agg_ap);
  run_edge(src_pa, dst_pa, h1_p, h1_a, att1_src + 1 * 256, att1_dst + 1 * 256, NP, NA, 256, agg_pa);
  run_edge(src_pp, dst_pp, h1_p, h1_p, att1_src + 2 * 256, att1_dst + 2 * 256, NP, NP, 256, agg_pp);
  run_edge(src_aa, dst_aa, h1_a, h1_a, att1_src + 3 * 256, att1_dst + 3 * 256, NA, NA, 256, agg_aa);

  // semantic attention (metapath order matches reference stacking: author=[pa,aa], paper=[ap,pp])
  float* res1_a = h1_a;   // h1 dead after aggregation: reuse in place
  float* res1_p = h1_p;
  run_sem(agg_pa, agg_aa, k1_W, k1_b, q1, NA, 256, res1_a);
  run_sem(agg_ap, agg_pp, k1_W, k1_b, q1, NP, 256, res1_p);

  // ---------------- layer 2 (author output only -> skip paper-destination work) ----------------
  gemm(res1_a, W2_a, b2_a, h2_a, NA, 256, 128);
  gemm(res1_p, W2_p, b2_p, h2_p, NP, 256, 128);

  run_edge(src_pa, dst_pa, h2_p, h2_a, att2_src + 1 * 128, att2_dst + 1 * 128, NP, NA, 128, agg2_pa);
  run_edge(src_aa, dst_aa, h2_a, h2_a, att2_src + 3 * 128, att2_dst + 3 * 128, NA, NA, 128, agg2_aa);

  run_sem(agg2_pa, agg2_aa, k2_W, k2_b, q2, NA, 128, (float*)d_out);
}
